// IDHead_59674275610746
// MI455X (gfx1250) — matrix operand order repaired and checked by reading, not compiled, not run on hardware
//
#include <hip/hip_runtime.h>
#include <hip/hip_bf16.h>

// ---------------- problem constants (from reference setup) ----------------
#define DIM     512
#define BQ      1024
#define NBANK   100000
#define TOPK    5
#define NSPLIT  25            // 100000 = 25 * 4000
#define NCHUNK  125           // 4000 / 32 chunks of 32 bank rows per split
#define MBLOCKS 64            // 1024 / 16 query-row blocks

typedef __bf16        bf16x16 __attribute__((ext_vector_type(16)));
typedef float         v8f     __attribute__((ext_vector_type(8)));
typedef unsigned int  u32x4   __attribute__((ext_vector_type(4)));
typedef int           i32x8   __attribute__((ext_vector_type(8)));
typedef int           i32x4   __attribute__((ext_vector_type(4)));

#define COMPILER_MEMFENCE() asm volatile("" ::: "memory")

__device__ __forceinline__ unsigned short f32_to_bf16_rne(float f) {
    unsigned int u = __float_as_uint(f);
    unsigned int r = (u + 0x7FFFu + ((u >> 16) & 1u)) >> 16;   // round-nearest-even
    return (unsigned short)r;
}

// -------------------------------------------------------------------------
// Kernel 1: row-wise L2 normalize + convert to bf16. One wave32 per row.
// -------------------------------------------------------------------------
__global__ __launch_bounds__(256) void norm_bf16_kernel(
    const float* __restrict__ in, unsigned short* __restrict__ out, int rows) {
    const int lane = threadIdx.x & 31;
    const int wave = threadIdx.x >> 5;
    const int row  = blockIdx.x * (blockDim.x >> 5) + wave;
    if (row >= rows) return;

    const float* p = in + (size_t)row * DIM;
    float v[16];
    float ss = 0.0f;
#pragma unroll
    for (int i = 0; i < 16; ++i) {
        v[i] = p[lane + 32 * i];
        ss += v[i] * v[i];
    }
#pragma unroll
    for (int off = 16; off > 0; off >>= 1)
        ss += __shfl_xor(ss, off, 32);

    const float inv = 1.0f / fmaxf(sqrtf(ss), 1e-12f);

    unsigned short* q = out + (size_t)row * DIM;
#pragma unroll
    for (int i = 0; i < 16; ++i)
        q[lane + 32 * i] = f32_to_bf16_rne(v[i] * inv);
}

// -------------------------------------------------------------------------
// TDM issue: DMA one 32-row x 1024B bank chunk (32KB) global -> LDS.
// D# per CDNA5 ISA ch.8: 2D tensor, data_size=8B units.
//   group0: count=1 | lds_addr | global_addr[56:0] | type=2
//   group1: data_size=3(8B); tensor_dim0=128, tensor_dim1=32;
//           tile_dim0=128, tile_dim1=32; tensor_dim0_stride=128
// This toolchain's builtin takes 6 args (g0 v4u, g1 v8i, v4i, v4i, v8i, cpol).
// -------------------------------------------------------------------------
__device__ __forceinline__ void tdm_load_chunk(const unsigned short* gsrc,
                                               unsigned lds_byte_addr) {
    const unsigned long long ga = (unsigned long long)(uintptr_t)gsrc;
    u32x4 g0;
    g0.x = 1u;                                        // count=1, load, user mode
    g0.y = lds_byte_addr;                             // [63:32]  lds_addr
    g0.z = (unsigned)(ga & 0xFFFFFFFFu);              // [95:64]  global_addr lo
    g0.w = (unsigned)((ga >> 32) & 0x1FFFFFFu)        // [120:96] global_addr hi
         | (2u << 30);                                // [127:126] type=2 (image)
    i32x8 g1;
    g1.s0 = (int)(3u << 16);        // workgroup_mask=0, data_size=3 (8B), no flags
    g1.s1 = (int)(128u << 16);      // atomic_barrier_addr=0 | tensor_dim0 lo16 = 128
    g1.s2 = (int)(32u << 16);       // tensor_dim0 hi16 = 0  | tensor_dim1 lo16 = 32
    g1.s3 = (int)(128u << 16);      // tensor_dim1 hi16 = 0  | tile_dim0 = 128
    g1.s4 = 32;                     // tile_dim1 = 32 rows   | tile_dim2 = 0
    g1.s5 = 128;                    // tensor_dim0_stride lo32 = 128 units
    g1.s6 = 0;                      // stride0 hi16 = 0 | tensor_dim1_stride lo16 = 0
    g1.s7 = 0;
    const i32x4 gz4 = {0, 0, 0, 0};            // groups 2/3 unused (2D)
    const i32x8 gz8 = {0, 0, 0, 0, 0, 0, 0, 0};
    __builtin_amdgcn_tensor_load_to_lds(g0, g1, gz4, gz4, gz8, 0);
}

// -------------------------------------------------------------------------
// Kernel 2: fused bf16 WMMA GEMM + per-row running top-5.
// grid = (64 M-blocks, 25 N-splits), block = 1 wave32.
// TDM double-buffers 32-row bank chunks into LDS (TENSORcnt pipelined);
// per chunk: two 16x16 tiles share the register-resident A fragments.
//
// A (16x32 bf16, lane = M, pair-interleave): halfs [kb..kb+7],[kb+16..kb+23]
//   of row M, kb = (lane<16 ? 0 : 8)                          (ISA 7.12.2)
// B (32x16 bf16, lane = N): halfs [kb..kb+7],[kb+16..kb+23] of row N, same kb as A
// C (16x16 f32): vgpr r, lane L -> M = r + (L>=16)*8, N = L%16
// -------------------------------------------------------------------------
__global__ __launch_bounds__(32) void simtopk_kernel(
    const unsigned short* __restrict__ zn,     // [1024][512] bf16 (normalized)
    const unsigned short* __restrict__ bn,     // [100000][512] bf16 (normalized)
    float* __restrict__ pvals, int* __restrict__ pidx) {
    __shared__ __align__(16) unsigned short bbuf[2][32 * DIM];  // 2 x 32KB
    __shared__ float smem[16 * 32];                             // C scatter

    const int lane   = threadIdx.x;
    const int mblock = blockIdx.x;             // 0..63
    const int split  = blockIdx.y;             // 0..24
    const int m      = lane & 15;
    const int hi     = lane >> 4;              // 0 or 1

    // wave-relative LDS byte offsets of the two buffers (low 32 bits of the
    // generic LDS pointer are the LDS address per the aperture mapping)
    const unsigned lbuf0 = (unsigned)(uintptr_t)(void*)&bbuf[0][0];
    const unsigned lbuf1 = (unsigned)(uintptr_t)(void*)&bbuf[1][0];

    // ---- load all 16 A fragments (covers K = 0..511) into registers ----
    bf16x16 afrag[16];
    {
        const unsigned short* arow = zn + (size_t)(mblock * 16 + m) * DIM + hi * 8;
#pragma unroll
        for (int s = 0; s < 16; ++s) {
            union { bf16x16 v; u32x4 u[2]; } t;
            t.u[0] = *(const u32x4*)(arow + s * 32);
            t.u[1] = *(const u32x4*)(arow + s * 32 + 16);
            afrag[s] = t.v;
        }
    }

    // ---- running top-5 (sorted descending), registers only ----
    float tv0 = -3.0e38f, tv1 = -3.0e38f, tv2 = -3.0e38f, tv3 = -3.0e38f, tv4 = -3.0e38f;
    int   ti0 = 0, ti1 = 0, ti2 = 0, ti3 = 0, ti4 = 0;

    const int nstart = split * (NCHUNK * 32);

    // prologue: start DMA of chunk 0
    tdm_load_chunk(bn + (size_t)nstart * DIM, lbuf0);

    for (int c = 0; c < NCHUNK; ++c) {
        const int nbase = nstart + c * 32;

        // issue next chunk's DMA, then wait for current chunk (<=1 in flight)
        if (c + 1 < NCHUNK) {
            tdm_load_chunk(bn + (size_t)(nbase + 32) * DIM,
                           ((c + 1) & 1) ? lbuf1 : lbuf0);
            __builtin_amdgcn_s_wait_tensorcnt(1);
        } else {
            __builtin_amdgcn_s_wait_tensorcnt(0);
        }
        COMPILER_MEMFENCE();

        const unsigned short* bb = &bbuf[c & 1][0];

        v8f acc0 = {};
        v8f acc1 = {};
#pragma unroll
        for (int s = 0; s < 16; ++s) {
            union { bf16x16 v; u32x4 u[2]; } b0, b1;
            const unsigned short* p0 = bb + (size_t)m * DIM + s * 32 + hi * 8;
            const unsigned short* p1 = p0 + (size_t)16 * DIM;
            b0.u[0] = *(const u32x4*)(p0);
            b0.u[1] = *(const u32x4*)(p0 + 16);
            b1.u[0] = *(const u32x4*)(p1);
            b1.u[1] = *(const u32x4*)(p1 + 16);
            acc0 = __builtin_amdgcn_wmma_f32_16x16x32_bf16(
                false, afrag[s], false, b0.v, (short)0, acc0, false, false);
            acc1 = __builtin_amdgcn_wmma_f32_16x16x32_bf16(
                false, afrag[s], false, b1.v, (short)0, acc1, false, false);
        }

        // scatter both C tiles to LDS per ISA layout
#pragma unroll
        for (int r = 0; r < 8; ++r) {
            smem[(r + hi * 8) * 32 + m]      = acc0[r];
            smem[(r + hi * 8) * 32 + 16 + m] = acc1[r];
        }
        __syncthreads();

        // all 32 lanes: lane handles row (lane&15), columns [hi*16, hi*16+16)
        {
            const int colbase = hi * 16;
#pragma unroll 4
            for (int j = 0; j < 16; ++j) {
                const float v = smem[m * 32 + colbase + j];
                if (v > tv4) {
                    const int idx = nbase + colbase + j;
                    if (v > tv0) {
                        tv4=tv3; ti4=ti3; tv3=tv2; ti3=ti2; tv2=tv1; ti2=ti1;
                        tv1=tv0; ti1=ti0; tv0=v;   ti0=idx;
                    } else if (v > tv1) {
                        tv4=tv3; ti4=ti3; tv3=tv2; ti3=ti2; tv2=tv1; ti2=ti1;
                        tv1=v;   ti1=idx;
                    } else if (v > tv2) {
                        tv4=tv3; ti4=ti3; tv3=tv2; ti3=ti2; tv2=v; ti2=idx;
                    } else if (v > tv3) {
                        tv4=tv3; ti4=ti3; tv3=v; ti3=idx;
                    } else {
                        tv4=v; ti4=idx;
                    }
                }
            }
        }
        __syncthreads();   // reconverged; EXEC all-ones again before next WMMA
    }

    // ---- write partial top-5 for this (mblock, split, half-wave) ----
    {
        const size_t base =
            (((size_t)(mblock * NSPLIT + split) * 2 + hi) * 16 + m) * TOPK;
        pvals[base + 0] = tv0; pidx[base + 0] = ti0;
        pvals[base + 1] = tv1; pidx[base + 1] = ti1;
        pvals[base + 2] = tv2; pidx[base + 2] = ti2;
        pvals[base + 3] = tv3; pidx[base + 3] = ti3;
        pvals[base + 4] = tv4; pidx[base + 4] = ti4;
    }
}

// -------------------------------------------------------------------------
// Kernel 3: merge 25*2 partial top-5 lists per query row -> final top-5,
// gather int64 labels, write [top_idx | top_vals | labels] as float.
// One wave32 per query row.
// -------------------------------------------------------------------------
__global__ __launch_bounds__(32) void merge_kernel(
    const float* __restrict__ pvals, const int* __restrict__ pidx,
    const long long* __restrict__ labels, float* __restrict__ out) {
    __shared__ float sv[256];
    __shared__ int   si[256];

    const int row    = blockIdx.x;
    const int lane   = threadIdx.x;
    const int mblock = row >> 4;
    const int r      = row & 15;

    for (int i = lane; i < 256; i += 32) { sv[i] = -3.0e38f; si[i] = 0; }
    __syncthreads();

    const int NCAND = NSPLIT * 2 * TOPK;   // 250
    for (int i = lane; i < NCAND; i += 32) {
        const int s   = i / (2 * TOPK);
        const int rem = i - s * (2 * TOPK);
        const int h   = rem / TOPK;
        const int t   = rem - h * TOPK;
        const size_t base =
            (((size_t)(mblock * NSPLIT + s) * 2 + h) * 16 + r) * TOPK + t;
        sv[i] = pvals[base];
        si[i] = pidx[base];
    }
    __syncthreads();

    float* out_idx = out;
    float* out_val = out + (size_t)BQ * TOPK;
    float* out_lab = out + (size_t)2 * BQ * TOPK;

    for (int t = 0; t < TOPK; ++t) {
        float bm = -3.0e38f; int bi = 0;
        for (int i = lane; i < 256; i += 32) {
            const float v = sv[i];
            if (v > bm) { bm = v; bi = i; }
        }
#pragma unroll
        for (int off = 16; off > 0; off >>= 1) {
            const float ov = __shfl_down(bm, off, 32);
            const int   oi = __shfl_down(bi, off, 32);
            if (ov > bm) { bm = ov; bi = oi; }
        }
        if (lane == 0) {
            const int gid = si[bi];
            out_idx[row * TOPK + t] = (float)gid;
            out_val[row * TOPK + t] = bm;
            out_lab[row * TOPK + t] = (float)labels[gid];
            sv[bi] = -3.0e38f;
        }
        __syncthreads();
    }
}

// -------------------------------------------------------------------------
extern "C" void kernel_launch(void* const* d_in, const int* in_sizes, int n_in,
                              void* d_out, int out_size, void* d_ws, size_t ws_size,
                              hipStream_t stream) {
    const float*     z      = (const float*)d_in[0];
    const float*     bank   = (const float*)d_in[1];
    const long long* labels = (const long long*)d_in[2];
    (void)in_sizes; (void)n_in; (void)out_size; (void)ws_size;

    // workspace layout (256B aligned)
    char* ws = (char*)d_ws;
    unsigned short* zn = (unsigned short*)ws;                       //   1,048,576 B
    unsigned short* bn = (unsigned short*)(ws + (1u << 20));        // 102,400,000 B
    float* pvals = (float*)(ws + 103448576u);                       //   1,024,000 B
    int*   pidx  = (int*)  (ws + 103448576u + 1024000u);            //   1,024,000 B

    // 1) normalize + bf16-convert queries and bank
    norm_bf16_kernel<<<dim3(BQ / 8),    256, 0, stream>>>(z,    zn, BQ);
    norm_bf16_kernel<<<dim3(NBANK / 8), 256, 0, stream>>>(bank, bn, NBANK);

    // 2) fused WMMA cosine-sim + per-split top-5 (TDM double-buffered)
    simtopk_kernel<<<dim3(MBLOCKS, NSPLIT), 32, 0, stream>>>(zn, bn, pvals, pidx);

    // 3) merge partials, gather labels, emit [idx | vals | labels]
    merge_kernel<<<dim3(BQ), 32, 0, stream>>>(pvals, pidx, labels, (float*)d_out);
}
